// EncoderLayer_60000693125405
// MI455X (gfx1250) — matrix operand order repaired and checked by reading, not compiled, not run on hardware
//
#include <hip/hip_runtime.h>
#include <cstdint>

#define D_MODEL 1024
#define NHEADS  16
#define DK      64
#define D_FF    4096
#define BATCH   4
#define SEQ     2048
#define ROWS    (BATCH * SEQ)   /* 8192 */
#define LN_EPS  1e-5f

typedef __attribute__((ext_vector_type(16))) __bf16 v16bf;
typedef __attribute__((ext_vector_type(8)))  float  v8f;
typedef __attribute__((ext_vector_type(4)))  int    v4i;

union FragU {
    uint4 q[2];
    v16bf v;
};

#define ZERO8 ((v8f){0.f,0.f,0.f,0.f,0.f,0.f,0.f,0.f})

__device__ __forceinline__ unsigned int bf16_bits(float a) {
    unsigned int u = __float_as_uint(a);
    return (u + 0x7FFFu + ((u >> 16) & 1u)) >> 16;   // round-to-nearest-even
}

__device__ __forceinline__ unsigned int pack2_bf16(float a, float b) {
    return (bf16_bits(b) << 16) | (bf16_bits(a) & 0xFFFFu);
}

__device__ __forceinline__ uint4 pack8(float4 a, float4 b) {
    return make_uint4(pack2_bf16(a.x, a.y), pack2_bf16(a.z, a.w),
                      pack2_bf16(b.x, b.y), pack2_bf16(b.z, b.w));
}

__device__ __forceinline__ v8f wmma_bf16(const FragU& a, const FragU& b, v8f c) {
    return __builtin_amdgcn_wmma_f32_16x16x32_bf16(
        false, a.v, false, b.v, (short)0, c, false, false);
}

// 16-byte global -> LDS copy on the CDNA5 async path
// (global_load_async_to_lds_b128, tracked by ASYNCcnt).
// Builtin prototype: void(v4i addrspace(1)*, v4i addrspace(3)*, Ii off, Ii cpol)
__device__ __forceinline__ void async_copy16(const void* g, void* l) {
#if __has_builtin(__builtin_amdgcn_global_load_async_to_lds_b128)
    __builtin_amdgcn_global_load_async_to_lds_b128(
        (__attribute__((address_space(1))) v4i*)g,
        (__attribute__((address_space(3))) v4i*)l, 0, 0);
#else
    *(uint4*)l = *(const uint4*)g;
#endif
}

__device__ __forceinline__ void wait_async() {
#if __has_builtin(__builtin_amdgcn_global_load_async_to_lds_b128)
#if __has_builtin(__builtin_amdgcn_s_wait_asynccnt)
    __builtin_amdgcn_s_wait_asynccnt(0);
#else
    asm volatile("s_wait_asynccnt 0" ::: "memory");
#endif
#endif
}

// ---------------------------------------------------------------------------
// Generic GEMM: C[M,N] = act(A[M,K] @ W[K,N] + bias[N]).
// act: 0=none, 1=exact GELU.  out_bf16: 0 -> f32 C, 1 -> bf16 C.
// 256 threads (8 waves), 128x128 tile, K-step 64 => 16 WMMA per barrier pair.
// ---------------------------------------------------------------------------
__global__ __launch_bounds__(256) void gemm_bias_act_kernel(
    const float* __restrict__ A, const float* __restrict__ W,
    const float* __restrict__ bias, void* __restrict__ Cout,
    int M, int N, int K, int act, int out_bf16)
{
    __shared__ __align__(16) unsigned int As[128][32];  // 128 x 64 bf16
    __shared__ __align__(16) unsigned int Bs[128][32];  // [n][k] 128 x 64 bf16

    const int t     = threadIdx.x;
    const int lane  = t & 31;
    const int wid   = t >> 5;
    const int m0    = blockIdx.y * 128;
    const int n0    = blockIdx.x * 128;
    const int waveM = (wid >> 1) * 32;
    const int waveN = (wid & 1) * 64;

    const int ar  = t >> 1;          // A loader: row 0..127
    const int acf = (t & 1) * 32;    // A loader: col 0 or 32 (floats)
    const int bn  = t & 127;         // B loader: output column
    const int brg = t >> 7;          // B loader: k-row group (0/1), 32 rows each

    const int fm = lane & 15;
    const int kh = lane >> 4;

    v8f acc[2][4];
#pragma unroll
    for (int i = 0; i < 2; ++i)
#pragma unroll
        for (int j = 0; j < 4; ++j) acc[i][j] = ZERO8;

    for (int k0 = 0; k0 < K; k0 += 64) {
        const float* ap = A + (size_t)(m0 + ar) * K + (k0 + acf);
#pragma unroll
        for (int i = 0; i < 4; ++i) {
            float4 x0 = *(const float4*)(ap + 8 * i);
            float4 x1 = *(const float4*)(ap + 8 * i + 4);
            *(uint4*)&As[ar][(acf >> 1) + 4 * i] = pack8(x0, x1);
        }

        const float* wp = W + (size_t)(k0 + brg * 32) * N + (n0 + bn);
        unsigned int wq[16];
#pragma unroll
        for (int i = 0; i < 16; ++i)
            wq[i] = pack2_bf16(wp[(size_t)(2 * i) * N], wp[(size_t)(2 * i + 1) * N]);
#pragma unroll
        for (int i = 0; i < 4; ++i)
            *(uint4*)&Bs[bn][brg * 16 + 4 * i] =
                make_uint4(wq[4 * i], wq[4 * i + 1], wq[4 * i + 2], wq[4 * i + 3]);

        if (k0 + 64 < K) {   // next K-tile -> global_prefetch_b8
            __builtin_prefetch(ap + 64, 0, 1);
            __builtin_prefetch(wp + (size_t)64 * N, 0, 1);
        }
        __syncthreads();

#pragma unroll
        for (int s = 0; s < 2; ++s) {
            FragU af[2], bfm[4];
#pragma unroll
            for (int mt = 0; mt < 2; ++mt) {
                const int row = waveM + mt * 16 + fm;
                af[mt].q[0] = *(const uint4*)&As[row][s * 16 + kh * 4];
                af[mt].q[1] = *(const uint4*)&As[row][s * 16 + 8 + kh * 4];
            }
#pragma unroll
            for (int nt = 0; nt < 4; ++nt) {
                const int col = waveN + nt * 16 + fm;
                bfm[nt].q[0] = *(const uint4*)&Bs[col][s * 16 + kh * 4];
                bfm[nt].q[1] = *(const uint4*)&Bs[col][s * 16 + 8 + kh * 4];
            }
#pragma unroll
            for (int mt = 0; mt < 2; ++mt)
#pragma unroll
                for (int nt = 0; nt < 4; ++nt)
                    acc[mt][nt] = wmma_bf16(af[mt], bfm[nt], acc[mt][nt]);
        }
        __syncthreads();
    }

#pragma unroll
    for (int mt = 0; mt < 2; ++mt) {
#pragma unroll
        for (int nt = 0; nt < 4; ++nt) {
            const int col   = n0 + waveN + nt * 16 + fm;
            const float bv  = bias[col];
            const int rbase = m0 + waveM + mt * 16 + kh * 8;
#pragma unroll
            for (int j = 0; j < 8; ++j) {
                float v = acc[mt][nt][j] + bv;
                if (act) v = 0.5f * v * (1.0f + erff(v * 0.70710678118654752f));
                const size_t idx = (size_t)(rbase + j) * N + col;
                if (out_bf16)
                    ((unsigned short*)Cout)[idx] = (unsigned short)bf16_bits(v);
                else
                    ((float*)Cout)[idx] = v;
            }
        }
    }
}

// ---------------------------------------------------------------------------
// scores[bh,q,k] = (Q[b,q,h,:] . K[b,k,h,:]) / 8.  Q,K are bf16 in memory,
// staged to LDS with async b128 copies (no conversion needed).
// ---------------------------------------------------------------------------
__global__ __launch_bounds__(256) void attn_scores_kernel(
    const unsigned short* __restrict__ Q, const unsigned short* __restrict__ Kb,
    float* __restrict__ attn)
{
    __shared__ __align__(16) unsigned int As[128][32];  // Q tile 128 x 64 bf16
    __shared__ __align__(16) unsigned int Bs[128][32];  // K tile 128 x 64 bf16

    const int t    = threadIdx.x;
    const int lane = t & 31;
    const int wid  = t >> 5;
    const int k0   = blockIdx.x * 128;
    const int q0   = blockIdx.y * 128;
    const int bh   = blockIdx.z;            // b*16 + h
    const int b    = bh >> 4;
    const int h    = bh & 15;
    const int waveM = (wid >> 1) * 32;
    const int waveN = (wid & 1) * 64;
    const int fm = lane & 15;
    const int kh = lane >> 4;

    const int r     = t >> 1;               // tile row 0..127
    const int halfB = (t & 1) * 64;         // byte offset within 128B row

    {
        const unsigned short* qrow =
            Q  + ((size_t)(b * SEQ + q0 + r)) * D_MODEL + h * DK;
        const unsigned short* krow =
            Kb + ((size_t)(b * SEQ + k0 + r)) * D_MODEL + h * DK;
#pragma unroll
        for (int c = 0; c < 4; ++c) {
            async_copy16((const char*)qrow + halfB + 16 * c,
                         (char*)&As[r][0] + halfB + 16 * c);
            async_copy16((const char*)krow + halfB + 16 * c,
                         (char*)&Bs[r][0] + halfB + 16 * c);
        }
    }
    wait_async();
    __syncthreads();

    v8f acc[2][4];
#pragma unroll
    for (int i = 0; i < 2; ++i)
#pragma unroll
        for (int j = 0; j < 4; ++j) acc[i][j] = ZERO8;

#pragma unroll
    for (int s = 0; s < 2; ++s) {           // two K-steps of 32 over d
        FragU af[2], bfm[4];
#pragma unroll
        for (int mt = 0; mt < 2; ++mt) {
            const int row = waveM + mt * 16 + fm;
            af[mt].q[0] = *(const uint4*)&As[row][s * 16 + kh * 4];
            af[mt].q[1] = *(const uint4*)&As[row][s * 16 + 8 + kh * 4];
        }
#pragma unroll
        for (int nt = 0; nt < 4; ++nt) {
            const int col = waveN + nt * 16 + fm;
            bfm[nt].q[0] = *(const uint4*)&Bs[col][s * 16 + kh * 4];
            bfm[nt].q[1] = *(const uint4*)&Bs[col][s * 16 + 8 + kh * 4];
        }
#pragma unroll
        for (int mt = 0; mt < 2; ++mt)
#pragma unroll
            for (int nt = 0; nt < 4; ++nt)
                acc[mt][nt] = wmma_bf16(af[mt], bfm[nt], acc[mt][nt]);
    }

    float* op = attn + ((size_t)bh * SEQ + q0) * SEQ + k0;
#pragma unroll
    for (int mt = 0; mt < 2; ++mt) {
#pragma unroll
        for (int nt = 0; nt < 4; ++nt) {
            const int col   = waveN + nt * 16 + fm;
            const int rbase = waveM + mt * 16 + kh * 8;
#pragma unroll
            for (int j = 0; j < 8; ++j)
                op[(size_t)(rbase + j) * SEQ + col] = acc[mt][nt][j] * 0.125f;
        }
    }
}

// ---------------------------------------------------------------------------
// Row softmax over 2048 elements, in place.
// ---------------------------------------------------------------------------
__global__ __launch_bounds__(256) void softmax_kernel(float* __restrict__ attn)
{
    __shared__ float red[256];
    const size_t row = blockIdx.x;
    float* p = attn + row * (size_t)SEQ;
    const int t = threadIdx.x;

    float4 a = *(float4*)(p + t * 8);
    float4 b = *(float4*)(p + t * 8 + 4);
    float m = fmaxf(fmaxf(fmaxf(a.x, a.y), fmaxf(a.z, a.w)),
                    fmaxf(fmaxf(b.x, b.y), fmaxf(b.z, b.w)));
    red[t] = m;
    __syncthreads();
    for (int s = 128; s > 0; s >>= 1) {
        if (t < s) red[t] = fmaxf(red[t], red[t + s]);
        __syncthreads();
    }
    const float rmax = red[0];
    __syncthreads();

    a.x = __expf(a.x - rmax); a.y = __expf(a.y - rmax);
    a.z = __expf(a.z - rmax); a.w = __expf(a.w - rmax);
    b.x = __expf(b.x - rmax); b.y = __expf(b.y - rmax);
    b.z = __expf(b.z - rmax); b.w = __expf(b.w - rmax);

    red[t] = a.x + a.y + a.z + a.w + b.x + b.y + b.z + b.w;
    __syncthreads();
    for (int s = 128; s > 0; s >>= 1) {
        if (t < s) red[t] += red[t + s];
        __syncthreads();
    }
    const float inv = 1.0f / red[0];

    a.x *= inv; a.y *= inv; a.z *= inv; a.w *= inv;
    b.x *= inv; b.y *= inv; b.z *= inv; b.w *= inv;
    *(float4*)(p + t * 8)     = a;
    *(float4*)(p + t * 8 + 4) = b;
}

// ---------------------------------------------------------------------------
// out[b,q,h,:] = sum_k P[bh,q,k] * V[b,k,h,:].  P is f32 (converted while
// staging to LDS, which also transposes V into [d][k]).
// ---------------------------------------------------------------------------
__global__ __launch_bounds__(256) void attn_pv_kernel(
    const float* __restrict__ attn, const float* __restrict__ V,
    float* __restrict__ out)
{
    __shared__ __align__(16) unsigned int As[128][16];  // P tile 128 x 32 bf16
    __shared__ __align__(16) unsigned int Bs[64][16];   // [d][k] 64 x 32 bf16

    const int t    = threadIdx.x;
    const int lane = t & 31;
    const int wid  = t >> 5;                // wave handles q rows wid*16..+15
    const int q0   = blockIdx.x * 128;
    const int bh   = blockIdx.y;
    const int b    = bh >> 4;
    const int h    = bh & 15;
    const int fm   = lane & 15;
    const int kh   = lane >> 4;

    const int ar = t >> 1;                  // P loader row
    const int ac = (t & 1) * 16;            // P loader col (floats)
    const int vn = t & 63;                  // V loader: d column
    const int vr = (t >> 6) * 8;            // V loader: k-row group base

    v8f acc[4];
#pragma unroll
    for (int j = 0; j < 4; ++j) acc[j] = ZERO8;

    const float* prow = attn + ((size_t)bh * SEQ + q0 + ar) * SEQ + ac;
    const float* vcol = V + ((size_t)(b * SEQ) + vr) * D_MODEL + h * DK + vn;

    for (int k0 = 0; k0 < SEQ; k0 += 32) {
        const float* ap = prow + k0;
        float4 a0 = *(const float4*)(ap + 0);
        float4 a1 = *(const float4*)(ap + 4);
        float4 a2 = *(const float4*)(ap + 8);
        float4 a3 = *(const float4*)(ap + 12);

        const float* vp = vcol + (size_t)k0 * D_MODEL;
        unsigned int vq[4];
#pragma unroll
        for (int i = 0; i < 4; ++i)
            vq[i] = pack2_bf16(vp[(size_t)(2 * i) * D_MODEL],
                               vp[(size_t)(2 * i + 1) * D_MODEL]);

        if (k0 + 32 < SEQ) {
            __builtin_prefetch(ap + 32, 0, 1);
            __builtin_prefetch(vp + (size_t)32 * D_MODEL, 0, 1);
        }

        *(uint4*)&As[ar][(ac >> 1)]     = pack8(a0, a1);
        *(uint4*)&As[ar][(ac >> 1) + 4] = pack8(a2, a3);
        *(uint4*)&Bs[vn][(vr >> 1)]     = make_uint4(vq[0], vq[1], vq[2], vq[3]);
        __syncthreads();

        FragU af, bfm[4];
        const int row = wid * 16 + fm;
        af.q[0] = *(const uint4*)&As[row][kh * 4];
        af.q[1] = *(const uint4*)&As[row][8 + kh * 4];
#pragma unroll
        for (int nt = 0; nt < 4; ++nt) {
            const int col = nt * 16 + fm;
            bfm[nt].q[0] = *(const uint4*)&Bs[col][kh * 4];
            bfm[nt].q[1] = *(const uint4*)&Bs[col][8 + kh * 4];
        }
#pragma unroll
        for (int nt = 0; nt < 4; ++nt)
            acc[nt] = wmma_bf16(af, bfm[nt], acc[nt]);
        __syncthreads();
    }

#pragma unroll
    for (int nt = 0; nt < 4; ++nt) {
        const int col   = h * DK + nt * 16 + fm;
        const int rbase = q0 + wid * 16 + kh * 8;
#pragma unroll
        for (int j = 0; j < 8; ++j)
            out[((size_t)(b * SEQ) + rbase + j) * D_MODEL + col] = acc[nt][j];
    }
}

// ---------------------------------------------------------------------------
// out[row,:] = LayerNorm(x[row,:] + r[row,:]) * gamma + beta.
// ---------------------------------------------------------------------------
__global__ __launch_bounds__(256) void add_layernorm_kernel(
    const float* __restrict__ x, const float* __restrict__ r,
    const float* __restrict__ gamma, const float* __restrict__ beta,
    float* __restrict__ out)
{
    __shared__ float red[256];
    const size_t row = blockIdx.x;
    const int t  = threadIdx.x;
    const int c0 = t * 4;

    float4 xv = *(const float4*)(x + row * D_MODEL + c0);
    float4 rv = *(const float4*)(r + row * D_MODEL + c0);
    float4 v  = {xv.x + rv.x, xv.y + rv.y, xv.z + rv.z, xv.w + rv.w};

    red[t] = v.x + v.y + v.z + v.w;
    __syncthreads();
    for (int s = 128; s > 0; s >>= 1) {
        if (t < s) red[t] += red[t + s];
        __syncthreads();
    }
    const float mu = red[0] * (1.0f / D_MODEL);
    __syncthreads();

    float4 d = {v.x - mu, v.y - mu, v.z - mu, v.w - mu};
    red[t] = d.x * d.x + d.y * d.y + d.z * d.z + d.w * d.w;
    __syncthreads();
    for (int s = 128; s > 0; s >>= 1) {
        if (t < s) red[t] += red[t + s];
        __syncthreads();
    }
    const float var = red[0] * (1.0f / D_MODEL);
    const float sc  = rsqrtf(var + LN_EPS);

    float4 g  = *(const float4*)(gamma + c0);
    float4 be = *(const float4*)(beta + c0);
    float4 o  = {d.x * sc * g.x + be.x, d.y * sc * g.y + be.y,
                 d.z * sc * g.z + be.z, d.w * sc * g.w + be.w};
    *(float4*)(out + row * D_MODEL + c0) = o;
}

// ---------------------------------------------------------------------------
extern "C" void kernel_launch(void* const* d_in, const int* in_sizes, int n_in,
                              void* d_out, int out_size, void* d_ws, size_t ws_size,
                              hipStream_t stream)
{
    (void)in_sizes; (void)n_in; (void)out_size; (void)ws_size;

    const float* x     = (const float*)d_in[0];
    const float* w_q   = (const float*)d_in[1];
    const float* b_q   = (const float*)d_in[2];
    const float* w_k   = (const float*)d_in[3];
    const float* b_k   = (const float*)d_in[4];
    const float* w_v   = (const float*)d_in[5];
    const float* b_v   = (const float*)d_in[6];
    const float* w_o   = (const float*)d_in[7];
    const float* b_o   = (const float*)d_in[8];
    const float* w_fc1 = (const float*)d_in[9];
    const float* b_fc1 = (const float*)d_in[10];
    const float* w_fc2 = (const float*)d_in[11];
    const float* b_fc2 = (const float*)d_in[12];
    const float* g1    = (const float*)d_in[13];
    const float* be1   = (const float*)d_in[14];
    const float* g2    = (const float*)d_in[15];
    const float* be2   = (const float*)d_in[16];

    // Output: y [B,S,D] then attn_weights [B,H,S,S], both f32.
    float* y    = (float*)d_out;
    float* attn = y + (size_t)ROWS * D_MODEL;

    // Workspace: 6 regions of ROWS*D_MODEL floats each (ffn1 spans 4).
    float* ws = (float*)d_ws;
    const size_t RD = (size_t)ROWS * D_MODEL;        // 8,388,608
    unsigned short* Qb16 = (unsigned short*)(ws);          // bf16 Q
    unsigned short* Kb16 = (unsigned short*)(ws + RD);     // bf16 K
    float* Vb  = ws + 2 * RD;
    float* AOb = ws + 3 * RD;
    float* Hb  = ws + 4 * RD;
    float* F1b = ws + 5 * RD;                        // ROWS x D_FF f32
    float* OPb = (float*)(ws);                       // reuse Q region (f32 fits)
    float* F2b = (float*)(ws + RD);                  // reuse K region

    dim3 blk(256);
    dim3 gD(D_MODEL / 128, ROWS / 128);              // (8, 64)
    dim3 gF(D_FF / 128, ROWS / 128);                 // (32, 64)

    // QKV projections (Q,K emitted as bf16 for async attention staging)
    gemm_bias_act_kernel<<<gD, blk, 0, stream>>>(x, w_q, b_q, Qb16, ROWS, D_MODEL, D_MODEL, 0, 1);
    gemm_bias_act_kernel<<<gD, blk, 0, stream>>>(x, w_k, b_k, Kb16, ROWS, D_MODEL, D_MODEL, 0, 1);
    gemm_bias_act_kernel<<<gD, blk, 0, stream>>>(x, w_v, b_v, Vb,   ROWS, D_MODEL, D_MODEL, 0, 0);

    // Attention
    attn_scores_kernel<<<dim3(SEQ / 128, SEQ / 128, BATCH * NHEADS), blk, 0, stream>>>(Qb16, Kb16, attn);
    softmax_kernel<<<dim3(BATCH * NHEADS * SEQ), blk, 0, stream>>>(attn);
    attn_pv_kernel<<<dim3(SEQ / 128, BATCH * NHEADS), blk, 0, stream>>>(attn, Vb, AOb);

    // Output projection + residual LN1
    gemm_bias_act_kernel<<<gD, blk, 0, stream>>>(AOb, w_o, b_o, OPb, ROWS, D_MODEL, D_MODEL, 0, 0);
    add_layernorm_kernel<<<dim3(ROWS), blk, 0, stream>>>(x, OPb, g1, be1, Hb);

    // FFN (GELU in FC1 epilogue) + residual LN2 -> y
    gemm_bias_act_kernel<<<gF, blk, 0, stream>>>(Hb, w_fc1, b_fc1, F1b, ROWS, D_FF, D_MODEL, 1, 0);
    gemm_bias_act_kernel<<<gD, blk, 0, stream>>>(F1b, w_fc2, b_fc2, F2b, ROWS, D_MODEL, D_FF, 0, 0);
    add_layernorm_kernel<<<dim3(ROWS), blk, 0, stream>>>(Hb, F2b, g2, be2, y);
}
